// LBLResNetBiLm_10419590660607
// MI455X (gfx1250) — matrix operand order repaired and checked by reading, not compiled, not run on hardware
//
#include <hip/hip_runtime.h>

#define Bsz 32
#define Sd  512
#define Hd  512
#define Wd  4
#define Ld  2
#define ROWS (Bsz * Sd)   // 16384

typedef __attribute__((ext_vector_type(16))) __bf16 v16bf;
typedef __attribute__((ext_vector_type(8)))  float  v8f;
typedef int v4i __attribute__((vector_size(16)));   // matches async builtin param

#if defined(__gfx1250__) && __has_builtin(__builtin_amdgcn_global_load_async_to_lds_b128)
#define HAVE_ASYNC 1
#else
#define HAVE_ASYNC 0
#endif

__device__ __forceinline__ unsigned short f32_to_bf16(float f) {
    unsigned int u = __float_as_uint(f);
    u += 0x7FFFu + ((u >> 16) & 1u);   // round-to-nearest-even
    return (unsigned short)(u >> 16);
}

#if HAVE_ASYNC
__device__ __forceinline__ void async_wait0() {
#if __has_builtin(__builtin_amdgcn_s_wait_asynccnt)
    __builtin_amdgcn_s_wait_asynccnt(0);
#else
    asm volatile("s_wait_asynccnt 0x0" ::: "memory");
#endif
}
#endif

// ---------------------------------------------------------------------------
// fp32 [L][K][N] -> bf16 transposed [L][N][K] (weights, once per launch)
// ---------------------------------------------------------------------------
__global__ void cvtT_kernel(const float* __restrict__ src,
                            unsigned short* __restrict__ dst)
{
    int s = blockIdx.x * blockDim.x + threadIdx.x;     // linear over L*H*H
    int n = s & (Hd - 1);
    int k = (s >> 9) & (Hd - 1);
    int l = s >> 18;
    dst[((size_t)l << 18) + ((size_t)n << 9) + k] = f32_to_bf16(src[s]);
}

// ---------------------------------------------------------------------------
// Windowed 5-tap combine: comb[b,t,:] = sum_k w[k] * padded[off + t + k, :]
// ---------------------------------------------------------------------------
__global__ void comb_kernel(const float* __restrict__ x, int xStride,
                            const float* __restrict__ pad,
                            const float* __restrict__ w5, int dir,
                            float* __restrict__ comb)
{
    int idx = blockIdx.x * blockDim.x + threadIdx.x;   // ROWS * (H/4) threads
    int h4 = (idx & 127) << 2;
    int t  = (idx >> 7) & (Sd - 1);
    int b  = idx >> 16;
    float4 acc = make_float4(0.f, 0.f, 0.f, 0.f);
#pragma unroll
    for (int k = 0; k <= Wd; ++k) {
        float wk = w5[k];
        int j = t + k;
        const float* src;
        if (dir == 0) {
            if (j < Wd) src = pad + (size_t)j * Hd + h4;
            else        src = x + ((size_t)b * Sd + (j - Wd)) * (size_t)xStride + h4;
        } else {
            if (j < Sd) src = x + ((size_t)b * Sd + j) * (size_t)xStride + h4;
            else        src = pad + (size_t)(j - Sd) * Hd + h4;
        }
        float4 v = *(const float4*)src;
        acc.x += wk * v.x; acc.y += wk * v.y; acc.z += wk * v.z; acc.w += wk * v.w;
    }
    *(float4*)(comb + ((size_t)(b * Sd + t) * Hd + h4)) = acc;
}

// ---------------------------------------------------------------------------
// LayerNorm (torch-unbiased std, eps added to std) -> bf16.  Wave32 per row.
// ---------------------------------------------------------------------------
__global__ void ln_kernel(const float* __restrict__ xin,
                          const float* __restrict__ g,
                          const float* __restrict__ bb,
                          unsigned short* __restrict__ aout)
{
    int wave = threadIdx.x >> 5;
    int lane = threadIdx.x & 31;
    int row  = blockIdx.x * 8 + wave;
    const float* xr = xin + (size_t)row * Hd;

    float4 v[4];
    float s = 0.f, q = 0.f;
#pragma unroll
    for (int c = 0; c < 4; ++c) {
        int col = c * 128 + lane * 4;
        v[c] = *(const float4*)(xr + col);
        s += v[c].x + v[c].y + v[c].z + v[c].w;
        q += v[c].x * v[c].x + v[c].y * v[c].y + v[c].z * v[c].z + v[c].w * v[c].w;
    }
#pragma unroll
    for (int off = 16; off > 0; off >>= 1) {
        s += __shfl_xor(s, off, 32);
        q += __shfl_xor(q, off, 32);
    }
    float mean = s * (1.0f / Hd);
    float var  = q * (1.0f / Hd) - mean * mean;
    var = fmaxf(var, 0.0f) * ((float)Hd / (float)(Hd - 1));
    float inv = 1.0f / (sqrtf(var) + 1e-6f);

    unsigned short* ar = aout + (size_t)row * Hd;
#pragma unroll
    for (int c = 0; c < 4; ++c) {
        int col = c * 128 + lane * 4;
        float4 gg = *(const float4*)(g + col);
        float4 bv = *(const float4*)(bb + col);
        ushort4 o;
        o.x = f32_to_bf16(gg.x * (v[c].x - mean) * inv + bv.x);
        o.y = f32_to_bf16(gg.y * (v[c].y - mean) * inv + bv.y);
        o.z = f32_to_bf16(gg.z * (v[c].z - mean) * inv + bv.z);
        o.w = f32_to_bf16(gg.w * (v[c].w - mean) * inv + bv.w);
        *(ushort4*)(ar + col) = o;
    }
}

// ---------------------------------------------------------------------------
// GEMM: C[16384,512] = A(bf16 [M][K]) x W (via Wt = bf16 [N][K]), f32 acc.
// Block tile 128x128, BK=32, double-buffered LDS, BOTH tiles staged with
// async global->LDS b128.  8 waves, each computing 32x64 (2x4 WMMAs).
// MODE 0: out = bf16(relu(acc+bias)), stride 512
// MODE 1: out = f32(acc+bias+resid), given stride
// ---------------------------------------------------------------------------
template <int MODE>
__global__ void gemm_kernel(const unsigned short* __restrict__ A,
                            const unsigned short* __restrict__ Bt,   // [N][K]
                            const float* __restrict__ bias,
                            const float* __restrict__ resid,
                            void* __restrict__ outp, int outStride)
{
    constexpr int BM = 128, BN = 128, BK = 32, AP = 40, BP = 40;
    __shared__ alignas(16) unsigned short ldsA[2][BM * AP];   // [m][k], padded
    __shared__ alignas(16) unsigned short ldsBT[2][BN * BP];  // [n][k], padded

    const int tid  = threadIdx.x;
    const int bm   = blockIdx.x >> 2;    // 512/BN = 4 tiles along N
    const int bn   = blockIdx.x & 3;
    const int lane = tid & 31;
    const int wave = tid >> 5;
    const int wm   = (wave >> 1) * 32;   // wave M offset (4 waves along M)
    const int wn   = (wave & 1) * 64;    // wave N offset (2 waves along N)
    const int mrow = lane & 15;
    const int hi   = lane >> 4;
    const int aKhi = hi * 8;             // A layout: lane-group K offset 8
    const int bKhi = hi * 16;            // unused; B loads use aKhi (K offset 8)
    const int bcol = lane & 15;

    // stage one 128x32 bf16 tile (A rows or Wt rows) into lds[buf]
    auto stage = [&](const unsigned short* __restrict__ gbase, int growBase,
                     unsigned short* ldsDst, int k0) {
#pragma unroll
        for (int it = 0; it < 2; ++it) {
            int chunk = tid + it * 256;         // 512 x 16B chunks
            int r  = chunk >> 2;
            int c8 = (chunk & 3) * 8;
            const unsigned short* gp = gbase + ((size_t)(growBase + r) * 512 + k0 + c8);
#if HAVE_ASYNC
            __builtin_amdgcn_global_load_async_to_lds_b128(
                (__attribute__((address_space(1))) v4i*)gp,
                (__attribute__((address_space(3))) v4i*)&ldsDst[r * AP + c8],
                0, 0);
#else
            *(uint4*)&ldsDst[r * AP + c8] = *(const uint4*)gp;
#endif
        }
    };

    v8f acc[2][4];
#pragma unroll
    for (int mt = 0; mt < 2; ++mt)
#pragma unroll
        for (int nt = 0; nt < 4; ++nt)
            acc[mt][nt] = (v8f){};

    stage(A,  bm * BM, ldsA[0],  0);
    stage(Bt, bn * BN, ldsBT[0], 0);
#if HAVE_ASYNC
    async_wait0();
#endif
    __syncthreads();

#pragma unroll 1
    for (int kk = 0; kk < 16; ++kk) {
        const int p = kk & 1;
        if (kk < 15) {                      // prefetch next tiles into buf p^1
            stage(A,  bm * BM, ldsA[p ^ 1],  (kk + 1) * BK);
            stage(Bt, bn * BN, ldsBT[p ^ 1], (kk + 1) * BK);
        }

        v16bf afrag[2];
#pragma unroll
        for (int mt = 0; mt < 2; ++mt)
#pragma unroll
            for (int v = 0; v < 8; ++v) {
                int kq = (v < 4 ? 2 * v : 16 + 2 * (v - 4)) + aKhi;
                ((unsigned int*)&afrag[mt])[v] =
                    *(const unsigned int*)&ldsA[p][(wm + mt * 16 + mrow) * AP + kq];
            }
#pragma unroll
        for (int nt = 0; nt < 4; ++nt) {
            v16bf bfrag;
#pragma unroll
            for (int v = 0; v < 8; ++v) {
                int kq = (v < 4 ? 2 * v : 16 + 2 * (v - 4)) + aKhi;
                ((unsigned int*)&bfrag)[v] =
                    *(const unsigned int*)&ldsBT[p][(wn + nt * 16 + bcol) * BP + kq];
            }
#pragma unroll
            for (int mt = 0; mt < 2; ++mt)
                acc[mt][nt] = __builtin_amdgcn_wmma_f32_16x16x32_bf16(
                    false, afrag[mt], false, bfrag,
                    (short)0, acc[mt][nt], false, false);
        }

#if HAVE_ASYNC
        async_wait0();                      // next-tile async writes landed
#endif
        __syncthreads();                    // everyone done reading buf p
    }

    // epilogue: C/D layout VGPR r -> (m = r + 8*hi, n = lane&15)
#pragma unroll
    for (int mt = 0; mt < 2; ++mt)
#pragma unroll
        for (int nt = 0; nt < 4; ++nt)
#pragma unroll
            for (int r = 0; r < 8; ++r) {
                int m = wm + mt * 16 + r + 8 * hi;
                int n = wn + nt * 16 + bcol;
                size_t grow = (size_t)(bm * BM + m);
                int gcol = bn * BN + n;
                float vv = acc[mt][nt][r] + bias[gcol];
                if (MODE == 0) {
                    vv = fmaxf(vv, 0.0f);
                    ((unsigned short*)outp)[grow * 512 + gcol] = f32_to_bf16(vv);
                } else {
                    vv += resid[grow * 512 + gcol];
                    ((float*)outp)[grow * (size_t)outStride + gcol] = vv;
                }
            }
}

// ---------------------------------------------------------------------------
extern "C" void kernel_launch(void* const* d_in, const int* in_sizes, int n_in,
                              void* d_out, int out_size, void* d_ws, size_t ws_size,
                              hipStream_t stream)
{
    const float* inputs = (const float*)d_in[0];
    // d_in[1] = masks (all ones, unused by reference)
    const float* fpads  = (const float*)d_in[2];
    const float* bpads  = (const float*)d_in[3];
    const float* fws    = (const float*)d_in[4];
    const float* bws    = (const float*)d_in[5];
    const float* f_ln_g = (const float*)d_in[6];
    const float* f_ln_b = (const float*)d_in[7];
    const float* b_ln_g = (const float*)d_in[8];
    const float* b_ln_b = (const float*)d_in[9];
    const float* f_w1   = (const float*)d_in[10];
    const float* f_b1   = (const float*)d_in[11];
    const float* f_w2   = (const float*)d_in[12];
    const float* f_b2   = (const float*)d_in[13];
    const float* b_w1   = (const float*)d_in[14];
    const float* b_b1   = (const float*)d_in[15];
    const float* b_w2   = (const float*)d_in[16];
    const float* b_b2   = (const float*)d_in[17];
    float* out = (float*)d_out;

    // ws: comb f32 (32MB) | A bf16 (16MB) | hidden bf16 (16MB) | Wt bf16 (4MB)
    float*          comb = (float*)d_ws;
    unsigned short* aBuf = (unsigned short*)((char*)d_ws + (size_t)ROWS * Hd * 4);
    unsigned short* h1   = aBuf + (size_t)ROWS * Hd;
    unsigned short* wf1  = h1 + (size_t)ROWS * Hd;
    unsigned short* wf2  = wf1 + (size_t)Ld * Hd * Hd;
    unsigned short* wb1  = wf2 + (size_t)Ld * Hd * Hd;
    unsigned short* wb2  = wb1 + (size_t)Ld * Hd * Hd;

    // one-time (per launch) weight convert + transpose: [L][K][N] -> [L][N][K]
    const int cvtBlocks = (Ld * Hd * Hd) / 256;   // 2048
    cvtT_kernel<<<cvtBlocks, 256, 0, stream>>>(f_w1, wf1);
    cvtT_kernel<<<cvtBlocks, 256, 0, stream>>>(f_w2, wf2);
    cvtT_kernel<<<cvtBlocks, 256, 0, stream>>>(b_w1, wb1);
    cvtT_kernel<<<cvtBlocks, 256, 0, stream>>>(b_w2, wb2);

    for (int i = 0; i < Ld; ++i) {
        for (int sb = 0; sb < 2; ++sb) {              // 0 = forward, 1 = backward
            const float* xb;
            int xs;
            if (i == 0) { xb = inputs; xs = Hd; }
            else {
                xb = out + (size_t)(i - 1) * ROWS * (2 * Hd) + (sb ? Hd : 0);
                xs = 2 * Hd;
            }
            const float* pad = (sb ? bpads : fpads) + (size_t)i * Wd * Hd;
            const float* w5  = (sb ? bws   : fws)   + (size_t)i * (Wd + 1);
            comb_kernel<<<ROWS * (Hd / 4) / 256, 256, 0, stream>>>(xb, xs, pad, w5, sb, comb);

            const float* g  = (sb ? b_ln_g : f_ln_g) + (size_t)i * Hd;
            const float* be = (sb ? b_ln_b : f_ln_b) + (size_t)i * Hd;
            ln_kernel<<<ROWS / 8, 256, 0, stream>>>(comb, g, be, aBuf);

            const unsigned short* w1 = (sb ? wb1 : wf1) + (size_t)i * Hd * Hd;
            const unsigned short* w2 = (sb ? wb2 : wf2) + (size_t)i * Hd * Hd;
            const float* b1 = (sb ? b_b1 : f_b1) + (size_t)i * Hd;
            const float* b2 = (sb ? b_b2 : f_b2) + (size_t)i * Hd;

            gemm_kernel<0><<<512, 256, 0, stream>>>(aBuf, w1, b1, nullptr, (void*)h1, 512);

            float* oslice = out + (size_t)i * ROWS * (2 * Hd) + (sb ? Hd : 0);
            gemm_kernel<1><<<512, 256, 0, stream>>>(h1, w2, b2, comb, (void*)oslice, 2 * Hd);
        }
    }
    (void)in_sizes; (void)n_in; (void)out_size; (void)ws_size;
}
